// Attention_82944408420466
// MI455X (gfx1250) — matrix operand order repaired and checked by reading, not compiled, not run on hardware
//
#include <hip/hip_runtime.h>

typedef __attribute__((ext_vector_type(16))) _Float16 v16h;
typedef __attribute__((ext_vector_type(8)))  float    v8f;

#define BB 16
#define HH 8
#define SS 512
#define DD 64
#define ROWSTRIDE 516   // 516 % 64 == 4 -> conflict-free column gathers

__device__ __forceinline__ float waveReduceMax(float v) {
#pragma unroll
  for (int off = 16; off >= 1; off >>= 1)
    v = fmaxf(v, __shfl_xor(v, off, 32));
  return v;
}
__device__ __forceinline__ float waveReduceSum(float v) {
#pragma unroll
  for (int off = 16; off >= 1; off >>= 1)
    v += __shfl_xor(v, off, 32);
  return v;
}

__global__ __launch_bounds__(256) void attn_dual_softmax_kernel(
    const float* __restrict__ Q, const float* __restrict__ K,
    const float* __restrict__ V, const float* __restrict__ gnn,
    const float* __restrict__ dist, const float* __restrict__ mask,
    float* __restrict__ out, float* __restrict__ outScores)
{
  __shared__ float sc[16][ROWSTRIDE];   // 33 KB: one 16x512 score row-block

  const int bid  = blockIdx.x;
  const int qt   = bid & 31;            // S/16 = 32 q-tiles
  const int bh   = bid >> 5;            // b*H + h
  const int b    = bh >> 3;             // H = 8
  const int q0   = qt * 16;

  const int tid  = threadIdx.x;
  const int w    = tid >> 5;            // wave id (wave32)
  const int lane = tid & 31;
  const int hi   = lane >> 4;
  const int lo   = lane & 15;

  const float* Qbh = Q + (size_t)bh * SS * DD;
  const float* Kbh = K + (size_t)bh * SS * DD;
  const float* Vbh = V + (size_t)bh * SS * DD;

  // ================= Phase 1: logits = QK^T/8 + mask*(-1e9) =================
  // A 16x32 in this code's contiguous k order: lane(lo)=M; lanes<16 hold K 0..15,
  // lanes>=16 hold K 16..31, packed as consecutive pairs per VGPR.
  v16h a0, a1;
  {
    const float* qrow = Qbh + (size_t)(q0 + lo) * DD;
    const int r1 = hi * 16;
    const int r2 = 8 + hi * 16;
#pragma unroll
    for (int j = 0; j < 8; ++j) {
      a0[j]     = (_Float16)qrow[r1 + j];
      a0[8 + j] = (_Float16)qrow[r2 + j];
      a1[j]     = (_Float16)qrow[32 + r1 + j];
      a1[8 + j] = (_Float16)qrow[32 + r2 + j];
    }
  }
  // Each wave computes 4 of the 32 16x16 logit tiles.
#pragma unroll
  for (int i = 0; i < 4; ++i) {
    const int n0 = (w * 4 + i) * 16;
    // B = K^T chunk (32x16): lane(lo)=N; lanes<16 hold K 0..15 of chunk,
    // lanes>=16 hold K 16..31 -> 16 consecutive floats of the key row.
    const float* krow = Kbh + (size_t)(n0 + lo) * DD;
    const int ks = hi * 16;
    v16h b0, b1;
#pragma unroll
    for (int j = 0; j < 16; ++j) {
      b0[j] = (_Float16)krow[ks + j];
      b1[j] = (_Float16)krow[32 + ks + j];
    }
    v8f acc = {};
    acc = __builtin_amdgcn_wmma_f32_16x16x32_f16(false, a0, false, b0,
                                                 (short)0, acc, false, false);
    acc = __builtin_amdgcn_wmma_f32_16x16x32_f16(false, a1, false, b1,
                                                 (short)0, acc, false, false);
    // C/D layout: VGPR r -> row r + 8*hi, col = n0 + lo
#pragma unroll
    for (int r = 0; r < 8; ++r) {
      const int m   = r + hi * 8;
      const int col = n0 + lo;
      // mask is streamed once per (q,k) element set -> non-temporal
      const float mv = __builtin_nontemporal_load(
          mask + (size_t)b * SS * SS + (size_t)(q0 + m) * SS + col);
      sc[m][col] = acc[r] * 0.125f + mv * (-1.0e9f);
    }
  }
  __syncthreads();

  // ======== Phase 2: softmax -> +dist+gnn -> softmax, emit scores ==========
  const size_t gBH = (size_t)bh * SS * SS;
#pragma unroll
  for (int rr = 0; rr < 2; ++rr) {
    const int row = w * 2 + rr;                 // 8 waves x 2 rows = 16
    const size_t gRow = gBH + (size_t)(q0 + row) * SS;
    float x[16];
    float m1 = -3.0e38f;
#pragma unroll
    for (int j = 0; j < 16; ++j) {
      x[j] = sc[row][lane + 32 * j];
      m1 = fmaxf(m1, x[j]);
    }
    m1 = waveReduceMax(m1);
    float s1 = 0.0f;
#pragma unroll
    for (int j = 0; j < 16; ++j) { x[j] = __expf(x[j] - m1); s1 += x[j]; }
    s1 = waveReduceSum(s1);
    const float inv1 = 1.0f / s1;
    float m2 = -3.0e38f;
#pragma unroll
    for (int j = 0; j < 16; ++j) {
      const int col = lane + 32 * j;
      // dist/gnn are read exactly once (268 MB stream): keep them out of L2
      // so the 32x-reused K/V tensors stay resident.
      const float t = x[j] * inv1 +
                      __builtin_nontemporal_load(dist + gRow + col) +
                      __builtin_nontemporal_load(gnn + gRow + col);
      x[j] = t;
      m2 = fmaxf(m2, t);
    }
    m2 = waveReduceMax(m2);
    float s2 = 0.0f;
#pragma unroll
    for (int j = 0; j < 16; ++j) { x[j] = __expf(x[j] - m2); s2 += x[j]; }
    s2 = waveReduceSum(s2);
    const float inv2 = 1.0f / s2;
#pragma unroll
    for (int j = 0; j < 16; ++j) {
      const int col = lane + 32 * j;
      const float s = x[j] * inv2;
      sc[row][col] = s;                                      // feed phase 3
      __builtin_nontemporal_store(s, outScores + gRow + col); // 134 MB stream
    }
  }
  __syncthreads();

  // =================== Phase 3: out = scores @ V ============================
  if (w < 4) {                                  // 4 waves x 16 cols = D
    const int colBase = w * 16;
    v8f acc = {};
    for (int kc = 0; kc < 16; ++kc) {           // K = 512 in chunks of 32
      const int kb = kc * 32;
      const int r1 = kb + hi * 16;
      const int r2 = kb + 8 + hi * 16;
      v16h a, bm;
#pragma unroll
      for (int j = 0; j < 8; ++j) {
        a[j]     = (_Float16)sc[lo][r1 + j];
        a[8 + j] = (_Float16)sc[lo][r2 + j];
      }
      const int ksv = kb + hi * 16;
      const float* vcol = Vbh + colBase + lo;
#pragma unroll
      for (int j = 0; j < 16; ++j)
        bm[j] = (_Float16)vcol[(size_t)(ksv + j) * DD];
      acc = __builtin_amdgcn_wmma_f32_16x16x32_f16(false, a, false, bm,
                                                   (short)0, acc, false, false);
    }
#pragma unroll
    for (int r = 0; r < 8; ++r) {
      const int m = r + hi * 8;
      __builtin_nontemporal_store(
          acc[r], out + ((size_t)bh * SS + q0 + m) * DD + colBase + lo);
    }
  }
}

extern "C" void kernel_launch(void* const* d_in, const int* in_sizes, int n_in,
                              void* d_out, int out_size, void* d_ws, size_t ws_size,
                              hipStream_t stream) {
  const float* Q    = (const float*)d_in[0];
  const float* K    = (const float*)d_in[1];
  const float* V    = (const float*)d_in[2];
  const float* gnn  = (const float*)d_in[3];
  const float* dist = (const float*)d_in[4];
  const float* mask = (const float*)d_in[5];
  float* out       = (float*)d_out;
  float* outScores = out + (size_t)BB * HH * SS * DD;  // tuple: (out, scores)

  const int blocks = BB * HH * (SS / 16);              // 4096
  attn_dual_softmax_kernel<<<blocks, 256, 0, stream>>>(
      Q, K, V, gnn, dist, mask, out, outScores);
}
